// TriangleMultiplicationOutgoing_1279900254331
// MI455X (gfx1250) — compile-verified, matrix operand order checked by reading, not run on hardware
//
#include <hip/hip_runtime.h>
#include <hip/hip_bf16.h>

#define NN 384
#define CZ 128
#define MROWS (NN * NN)  // 147456

typedef __attribute__((ext_vector_type(16))) __bf16 v16bf;
typedef __attribute__((ext_vector_type(8)))  __bf16 v8bf;
typedef __attribute__((ext_vector_type(8)))  float  v8f;
typedef __attribute__((ext_vector_type(4)))  float  v4f;
typedef __attribute__((ext_vector_type(4)))  int    v4i;

typedef __attribute__((address_space(1))) v4i gv4i;
typedef __attribute__((address_space(3))) v4i lv4i;

#if defined(__gfx1250__) && __has_builtin(__builtin_amdgcn_global_load_async_to_lds_b128)
#define USE_ASYNC_LDS 1
#else
#define USE_ASYNC_LDS 0
#endif

#if __has_builtin(__builtin_amdgcn_s_wait_asynccnt)
#define WAIT_ASYNCCNT0() __builtin_amdgcn_s_wait_asynccnt(0)
#else
#define WAIT_ASYNCCNT0() asm volatile("s_wait_asynccnt 0x0" ::: "memory")
#endif

__device__ __forceinline__ float sigmoidf_(float x) {
    return 1.0f / (1.0f + __expf(-x));
}

__device__ __forceinline__ v16bf pack16(v8bf lo, v8bf hi) {
    return __builtin_shufflevector(lo, hi, 0,1,2,3,4,5,6,7,8,9,10,11,12,13,14,15);
}

// 16-byte global -> LDS copy. Async (ASYNCcnt-tracked, VGPR-bypassing) when the
// toolchain exposes the gfx1250 builtin, synchronous v8bf copy otherwise.
__device__ __forceinline__ void copy16_g2l(__bf16* lds, const __bf16* g) {
#if USE_ASYNC_LDS
    __builtin_amdgcn_global_load_async_to_lds_b128(
        (gv4i*)(uintptr_t)g,
        (lv4i*)(uint32_t)(uintptr_t)lds,
        0, 0);
#else
    *(v8bf*)lds = *(const v8bf*)g;
#endif
}

// ---------------------------------------------------------------------------
// Kernel 0: convert the 6 [128x128] fp32 weight matrices to bf16 (row-major).
// Order in wb: 0=g1_w 1=g2_w 2=l1_w 3=l2_w 4=gate_w 5=out_w
// ---------------------------------------------------------------------------
__global__ __launch_bounds__(256) void convert_weights_kernel(
    const float* __restrict__ g1, const float* __restrict__ g2,
    const float* __restrict__ l1, const float* __restrict__ l2,
    const float* __restrict__ gw, const float* __restrict__ ow,
    __bf16* __restrict__ wb)
{
    int idx = blockIdx.x * 256 + threadIdx.x;   // 0 .. 98303
    int w = idx >> 14;                          // /16384
    int e = idx & 16383;
    const float* srcs[6] = {g1, g2, l1, l2, gw, ow};
    wb[idx] = (__bf16)srcs[w][e];
}

// ---------------------------------------------------------------------------
// Kernel 1: fused LayerNorm(pair) + 5 projections (g1,l1,g2,l2,gate) via WMMA.
//   Abuf[c][m] = sigmoid(z@g1^T+b)*(z@l1^T+b)   (bf16, channel-major)
//   Bbuf[c][m] = sigmoid(z@g2^T+b)*(z@l2^T+b)   (bf16, channel-major)
//   Gbuf[m][c] = sigmoid(z@gate^T+b)            (bf16, natural layout)
// One block (128 thr = 4 waves) handles 16 rows of z.
// ---------------------------------------------------------------------------
__global__ __launch_bounds__(128) void ln_proj_kernel(
    const float* __restrict__ pair,
    const float* __restrict__ lnw, const float* __restrict__ lnb,
    const __bf16* __restrict__ wb,
    const float* __restrict__ g1b, const float* __restrict__ g2b,
    const float* __restrict__ l1b, const float* __restrict__ l2b,
    const float* __restrict__ gtb,
    __bf16* __restrict__ Abuf, __bf16* __restrict__ Bbuf,
    __bf16* __restrict__ Gbuf)
{
    __shared__ __bf16 zs[16][136];   // padded: 272B row stride, 16B aligned

    const int t  = threadIdx.x;
    const int m0 = blockIdx.x * 16;

    // ---- LayerNorm phase: 8 lanes per row, 16 channels per lane ----
    {
        const int row = t >> 3;
        const int seg = t & 7;
        const float* src = pair + (size_t)(m0 + row) * CZ + seg * 16;
        float x[16];
        #pragma unroll
        for (int i = 0; i < 4; ++i) {
            v4f v = *(const v4f*)(src + i * 4);
            x[4*i+0] = v[0]; x[4*i+1] = v[1]; x[4*i+2] = v[2]; x[4*i+3] = v[3];
        }
        float s = 0.f, s2 = 0.f;
        #pragma unroll
        for (int i = 0; i < 16; ++i) { s += x[i]; s2 += x[i] * x[i]; }
        #pragma unroll
        for (int m = 1; m < 8; m <<= 1) {
            s  += __shfl_xor(s,  m, 8);
            s2 += __shfl_xor(s2, m, 8);
        }
        float mean = s * (1.0f / 128.0f);
        float var  = s2 * (1.0f / 128.0f) - mean * mean;
        float rs   = __frsqrt_rn(var + 1e-5f);
        #pragma unroll
        for (int i = 0; i < 16; ++i) {
            int c = seg * 16 + i;
            float zn = (x[i] - mean) * rs * lnw[c] + lnb[c];
            zs[row][c] = (__bf16)zn;
        }
    }
    __syncthreads();

    // ---- WMMA phase: wave handles output-column tiles {wave, wave+4} ----
    const int lane = t & 31;
    const int wave = t >> 5;
    const int lm = lane & 15;
    const int kh = lane >> 4;

    for (int nt = wave; nt < 8; nt += 4) {
        const int nb = nt * 16;
        v8f acc[5] = {};
        #pragma unroll
        for (int k0 = 0; k0 < 128; k0 += 32) {
            const __bf16* ap = &zs[lm][k0 + kh * 8];
            v16bf a = pack16(*(const v8bf*)ap, *(const v8bf*)(ap + 16));
            #pragma unroll
            for (int w = 0; w < 5; ++w) {
                const __bf16* bp = wb + w * 16384 + (nb + lm) * 128 + k0 + kh * 8;
                v16bf b = pack16(*(const v8bf*)bp, *(const v8bf*)(bp + 16));
                acc[w] = __builtin_amdgcn_wmma_f32_16x16x32_bf16(
                    false, a, false, b, (short)0, acc[w], false, false);
            }
        }
        const int n = nb + lm;
        const float bg1 = g1b[n], bg2 = g2b[n], bl1 = l1b[n], bl2 = l2b[n], bgt = gtb[n];
        v8bf av, bv;
        #pragma unroll
        for (int r = 0; r < 8; ++r) {
            float aval = sigmoidf_(acc[0][r] + bg1) * (acc[2][r] + bl1);
            float bval = sigmoidf_(acc[1][r] + bg2) * (acc[3][r] + bl2);
            float gval = sigmoidf_(acc[4][r] + bgt);
            av[r] = (__bf16)aval;
            bv[r] = (__bf16)bval;
            int m = m0 + kh * 8 + r;
            Gbuf[(size_t)m * CZ + n] = (__bf16)gval;
        }
        // rows r..r+7 are contiguous in m -> one 16B store per lane
        size_t moff = (size_t)n * MROWS + m0 + kh * 8;
        *(v8bf*)(Abuf + moff) = av;
        *(v8bf*)(Bbuf + moff) = bv;
    }
}

// ---------------------------------------------------------------------------
// Kernel 2: triangle einsum as 128 batched NT-GEMMs (one channel per block.z).
// vals[c][i][j] = sum_k A_c[i][k] * B_c[j][k].  64x64 output tile per block,
// 2x2 16x16 WMMA tiles per wave.  K is staged through DOUBLE-BUFFERED LDS
// (80B row stride: bank-conflict-free for the 16-row b128 gather pattern);
// tile k+1 is fetched with global_load_async_to_lds_b128 (ASYNCcnt) while
// tile k runs its WMMAs, then one s_wait_asynccnt + barrier per iteration.
// ---------------------------------------------------------------------------
__global__ __launch_bounds__(128) void tri_gemm_kernel(
    const __bf16* __restrict__ Abuf, const __bf16* __restrict__ Bbuf,
    float* __restrict__ vals)
{
    __shared__ __bf16 As[2][64 * 40];
    __shared__ __bf16 Bs[2][64 * 40];

    const int t  = threadIdx.x;
    const int c  = blockIdx.z;
    const int i0 = blockIdx.x * 64;
    const int j0 = blockIdx.y * 64;
    const __bf16* Ac = Abuf + (size_t)c * MROWS;
    const __bf16* Bc = Bbuf + (size_t)c * MROWS;

    const int lane = t & 31;
    const int wave = t >> 5;
    const int wi = (wave >> 1) * 32;
    const int wj = (wave & 1) * 32;
    const int lm = lane & 15;
    const int kh = lane >> 4;

    const int lrow  = t >> 1;         // loader row 0..63
    const int lhalf = (t & 1) * 16;   // 16 bf16 per loader thread

    v8f acc[2][2] = {};

    const __bf16* gA = Ac + (size_t)(i0 + lrow) * NN + lhalf;
    const __bf16* gB = Bc + (size_t)(j0 + lrow) * NN + lhalf;
    const int ldsOff = lrow * 40 + lhalf;

    // prologue: fetch k-tile 0 into buffer 0
    copy16_g2l(&As[0][ldsOff],     gA);
    copy16_g2l(&As[0][ldsOff + 8], gA + 8);
    copy16_g2l(&Bs[0][ldsOff],     gB);
    copy16_g2l(&Bs[0][ldsOff + 8], gB + 8);
#if USE_ASYNC_LDS
    WAIT_ASYNCCNT0();
#endif
    __syncthreads();

    const int KT = NN / 32;  // 12
    for (int kt = 0; kt < KT; ++kt) {
        const int buf = kt & 1;

        // issue async fetch of the next k-tile into the other buffer
        if (kt + 1 < KT) {
            const int k1 = (kt + 1) * 32;
            copy16_g2l(&As[buf ^ 1][ldsOff],     gA + k1);
            copy16_g2l(&As[buf ^ 1][ldsOff + 8], gA + k1 + 8);
            copy16_g2l(&Bs[buf ^ 1][ldsOff],     gB + k1);
            copy16_g2l(&Bs[buf ^ 1][ldsOff + 8], gB + k1 + 8);
        }

        // compute on the current buffer
        v16bf a[2], b[2];
        #pragma unroll
        for (int mi = 0; mi < 2; ++mi) {
            const __bf16* p = &As[buf][(wi + mi * 16 + lm) * 40 + kh * 8];
            a[mi] = pack16(*(const v8bf*)p, *(const v8bf*)(p + 16));
        }
        #pragma unroll
        for (int nj = 0; nj < 2; ++nj) {
            const __bf16* p = &Bs[buf][(wj + nj * 16 + lm) * 40 + kh * 8];
            b[nj] = pack16(*(const v8bf*)p, *(const v8bf*)(p + 16));
        }
        #pragma unroll
        for (int mi = 0; mi < 2; ++mi)
            #pragma unroll
            for (int nj = 0; nj < 2; ++nj)
                acc[mi][nj] = __builtin_amdgcn_wmma_f32_16x16x32_bf16(
                    false, a[mi], false, b[nj], (short)0, acc[mi][nj], false, false);

        if (kt + 1 < KT) {
#if USE_ASYNC_LDS
            WAIT_ASYNCCNT0();
#endif
            __syncthreads();
        }
    }

    float* vc = vals + (size_t)c * MROWS;
    #pragma unroll
    for (int mi = 0; mi < 2; ++mi)
        #pragma unroll
        for (int nj = 0; nj < 2; ++nj) {
            int jj = j0 + wj + nj * 16 + lm;
            #pragma unroll
            for (int r = 0; r < 8; ++r) {
                int ii = i0 + wi + mi * 16 + kh * 8 + r;
                vc[(size_t)ii * NN + jj] = acc[mi][nj][r];
            }
        }
}

// ---------------------------------------------------------------------------
// Kernel 3: final LayerNorm(vals) + @out_w^T + bias, gated by sigmoid G.
// vals is [c][ij] so the per-channel gather is fully coalesced.
// ---------------------------------------------------------------------------
__global__ __launch_bounds__(128) void out_kernel(
    const float* __restrict__ vals, const __bf16* __restrict__ Gbuf,
    const __bf16* __restrict__ wb,   // +5*16384 = out_w bf16
    const float* __restrict__ lnw, const float* __restrict__ lnb,
    const float* __restrict__ outb,
    float* __restrict__ out)
{
    __shared__ float  vs[16][132];
    __shared__ __bf16 zs[16][136];

    const int t  = threadIdx.x;
    const int m0 = blockIdx.x * 16;

    // gather 16 rows x 128 channels (coalesced: contiguous ij per channel)
    {
        const float* src = vals + (size_t)t * MROWS + m0;
        #pragma unroll
        for (int r = 0; r < 16; r += 4) {
            v4f v = *(const v4f*)(src + r);
            vs[r + 0][t] = v[0]; vs[r + 1][t] = v[1];
            vs[r + 2][t] = v[2]; vs[r + 3][t] = v[3];
        }
    }
    __syncthreads();

    // LayerNorm
    {
        const int row = t >> 3;
        const int seg = t & 7;
        float x[16];
        #pragma unroll
        for (int i = 0; i < 16; ++i) x[i] = vs[row][seg * 16 + i];
        float s = 0.f, s2 = 0.f;
        #pragma unroll
        for (int i = 0; i < 16; ++i) { s += x[i]; s2 += x[i] * x[i]; }
        #pragma unroll
        for (int m = 1; m < 8; m <<= 1) {
            s  += __shfl_xor(s,  m, 8);
            s2 += __shfl_xor(s2, m, 8);
        }
        float mean = s * (1.0f / 128.0f);
        float var  = s2 * (1.0f / 128.0f) - mean * mean;
        float rs   = __frsqrt_rn(var + 1e-5f);
        #pragma unroll
        for (int i = 0; i < 16; ++i) {
            int c = seg * 16 + i;
            zs[row][c] = (__bf16)((x[i] - mean) * rs * lnw[c] + lnb[c]);
        }
    }
    __syncthreads();

    // WMMA with out_w, then gate and store
    const int lane = t & 31;
    const int wave = t >> 5;
    const int lm = lane & 15;
    const int kh = lane >> 4;
    const __bf16* wout = wb + 5 * 16384;

    for (int nt = wave; nt < 8; nt += 4) {
        const int nb = nt * 16;
        v8f acc = {};
        #pragma unroll
        for (int k0 = 0; k0 < 128; k0 += 32) {
            const __bf16* ap = &zs[lm][k0 + kh * 8];
            v16bf a = pack16(*(const v8bf*)ap, *(const v8bf*)(ap + 16));
            const __bf16* bp = wout + (nb + lm) * 128 + k0 + kh * 8;
            v16bf b = pack16(*(const v8bf*)bp, *(const v8bf*)(bp + 16));
            acc = __builtin_amdgcn_wmma_f32_16x16x32_bf16(
                false, a, false, b, (short)0, acc, false, false);
        }
        const int n = nb + lm;
        const float bias = outb[n];
        #pragma unroll
        for (int r = 0; r < 8; ++r) {
            int m = m0 + kh * 8 + r;
            float g = (float)Gbuf[(size_t)m * CZ + n];
            out[(size_t)m * CZ + n] = g * (acc[r] + bias);
        }
    }
}

// ---------------------------------------------------------------------------
extern "C" void kernel_launch(void* const* d_in, const int* in_sizes, int n_in,
                              void* d_out, int out_size, void* d_ws, size_t ws_size,
                              hipStream_t stream) {
    const float* pair   = (const float*)d_in[0];
    const float* lnpw   = (const float*)d_in[1];
    const float* lnpb   = (const float*)d_in[2];
    const float* lnow   = (const float*)d_in[3];
    const float* lnob   = (const float*)d_in[4];
    const float* g1w    = (const float*)d_in[5];
    const float* g1b    = (const float*)d_in[6];
    const float* g2w    = (const float*)d_in[7];
    const float* g2b    = (const float*)d_in[8];
    const float* l1w    = (const float*)d_in[9];
    const float* l1b    = (const float*)d_in[10];
    const float* l2w    = (const float*)d_in[11];
    const float* l2b    = (const float*)d_in[12];
    const float* gatew  = (const float*)d_in[13];
    const float* gateb  = (const float*)d_in[14];
    const float* outw   = (const float*)d_in[15];
    const float* outb   = (const float*)d_in[16];

    char* ws = (char*)d_ws;
    __bf16* wb   = (__bf16*)ws;                         // 6*16384 bf16
    __bf16* Abuf = (__bf16*)(ws + 262144);              // [128][147456] bf16
    __bf16* Bbuf = Abuf + (size_t)CZ * MROWS;
    __bf16* Gbuf = Bbuf + (size_t)CZ * MROWS;           // [147456][128] bf16
    float*  vals = (float*)(ws + 262144 + 3ull * 37748736ull);  // [128][147456] f32

    convert_weights_kernel<<<384, 256, 0, stream>>>(g1w, g2w, l1w, l2w, gatew, outw, wb);
    ln_proj_kernel<<<MROWS / 16, 128, 0, stream>>>(
        pair, lnpw, lnpb, wb, g1b, g2b, l1b, l2b, gateb, Abuf, Bbuf, Gbuf);
    tri_gemm_kernel<<<dim3(NN / 64, NN / 64, CZ), 128, 0, stream>>>(Abuf, Bbuf, vals);
    out_kernel<<<MROWS / 16, 128, 0, stream>>>(
        vals, Gbuf, wb, lnow, lnob, outb, (float*)d_out);
}
